// EfficientMultiHeadAttention_23390391894066
// MI455X (gfx1250) — compile-verified, matrix operand order checked by reading, not run on hardware
//
#include <hip/hip_runtime.h>

// ---------------------------------------------------------------------------
// Multi-head attention for MI455X (gfx1250, wave32, WMMA).
// All GEMMs run through v_wmma_f32_16x16x32_bf16. Attention probabilities are
// materialized in f32 directly into d_out (required output #2).
// ---------------------------------------------------------------------------

typedef __attribute__((ext_vector_type(16))) __bf16 v16bf;
typedef __attribute__((ext_vector_type(8)))  __bf16 v8bf;
typedef __attribute__((ext_vector_type(8)))  float  v8f;
typedef __attribute__((ext_vector_type(4)))  float  v4f;

#define BB   2
#define HH   16
#define SS   2048
#define DD   1024
#define DH   64
#define MS   4096   // B*S
#define NEGI (-1e6f)

union FragU {
  v16bf v;
  v8bf  h[2];
  __bf16 e[16];
};

// A/B fragment for 16x32 bf16 tile, operand stored row-major [row][K].
// p must already point at: base + (row0 + (lane&15))*ld + k0 + (lane>>4)*8
__device__ __forceinline__ v16bf frag_bf16(const __bf16* __restrict__ p) {
  FragU f;
  f.h[0] = *(const v8bf*)(p);
  f.h[1] = *(const v8bf*)(p + 16);
  return f.v;
}

// Same fragment, but source is f32 (attention probs) converted on the fly.
__device__ __forceinline__ v16bf frag_f32(const float* __restrict__ p) {
  v4f a = *(const v4f*)(p);
  v4f b = *(const v4f*)(p + 4);
  v4f c = *(const v4f*)(p + 16);
  v4f d = *(const v4f*)(p + 20);
  FragU f;
#pragma unroll
  for (int i = 0; i < 4; ++i) {
    f.e[i]      = (__bf16)a[i];
    f.e[4 + i]  = (__bf16)b[i];
    f.e[8 + i]  = (__bf16)c[i];
    f.e[12 + i] = (__bf16)d[i];
  }
  return f.v;
}

__device__ __forceinline__ v8f wmma_bf16(v16bf a, v16bf b, v8f c) {
  return __builtin_amdgcn_wmma_f32_16x16x32_bf16(
      false, a, false, b, (short)0, c, false, false);
}

// ---------------------------------------------------------------------------
// f32 -> bf16 conversion
// ---------------------------------------------------------------------------
__global__ __launch_bounds__(256) void mha_cvt_bf16(
    const float* __restrict__ src, __bf16* __restrict__ dst, int n) {
  int i = blockIdx.x * 256 + threadIdx.x;
  if (i < n) dst[i] = (__bf16)src[i];
}

// ---------------------------------------------------------------------------
// Projection GEMM: C[M,N] = A[M,K] @ W[N,K]^T + bias[N]
// Block: 128x128 tile of C, 256 threads = 8 waves, wave tile 32(M) x 64(N).
// MODE 0: store bf16, head layout   Qh[((b*H+h)*S + s)*64 + d]
// MODE 1: store bf16, head transp.  VhT[((b*H+h)*64 + d)*S + s]
// MODE 2: store f32,  plain         out[m*N + n]
// ---------------------------------------------------------------------------
template <int MODE>
__global__ __launch_bounds__(256) void proj_gemm(
    const __bf16* __restrict__ A, const __bf16* __restrict__ W,
    const float* __restrict__ bias, void* __restrict__ dst,
    int M, int N, int K) {
  const int tid  = threadIdx.x;
  const int lane = tid & 31;
  const int wave = tid >> 5;
  const int wm   = wave >> 1;          // 0..3
  const int wn   = wave & 1;           // 0..1
  const int row0 = blockIdx.y * 128 + wm * 32;
  const int col0 = blockIdx.x * 128 + wn * 64;
  const int r    = lane & 15;
  const int hi   = lane >> 4;

  v8f acc[2][4];
#pragma unroll
  for (int i = 0; i < 2; ++i)
#pragma unroll
    for (int j = 0; j < 4; ++j) acc[i][j] = (v8f){0.f,0.f,0.f,0.f,0.f,0.f,0.f,0.f};

  const __bf16* ap0 = A + (size_t)(row0 + r) * K + hi * 8;
  const __bf16* ap1 = ap0 + (size_t)16 * K;
  const __bf16* bp0 = W + (size_t)(col0 + r) * K + hi * 8;

  for (int k0 = 0; k0 < K; k0 += 32) {
    v16bf a0 = frag_bf16(ap0 + k0);
    v16bf a1 = frag_bf16(ap1 + k0);
    v16bf b0 = frag_bf16(bp0 + k0);
    v16bf b1 = frag_bf16(bp0 + (size_t)16 * K + k0);
    v16bf b2 = frag_bf16(bp0 + (size_t)32 * K + k0);
    v16bf b3 = frag_bf16(bp0 + (size_t)48 * K + k0);
    acc[0][0] = wmma_bf16(a0, b0, acc[0][0]);
    acc[0][1] = wmma_bf16(a0, b1, acc[0][1]);
    acc[0][2] = wmma_bf16(a0, b2, acc[0][2]);
    acc[0][3] = wmma_bf16(a0, b3, acc[0][3]);
    acc[1][0] = wmma_bf16(a1, b0, acc[1][0]);
    acc[1][1] = wmma_bf16(a1, b1, acc[1][1]);
    acc[1][2] = wmma_bf16(a1, b2, acc[1][2]);
    acc[1][3] = wmma_bf16(a1, b3, acc[1][3]);
  }

#pragma unroll
  for (int i = 0; i < 2; ++i)
#pragma unroll
    for (int j = 0; j < 4; ++j)
#pragma unroll
      for (int e = 0; e < 8; ++e) {
        int rr = row0 + i * 16 + hi * 8 + e;   // m index
        int cc = col0 + j * 16 + r;            // n index
        float val = acc[i][j][e] + bias[cc];
        if (MODE == 2) {
          ((float*)dst)[(size_t)rr * N + cc] = val;
        } else {
          int b = rr >> 11, s = rr & (SS - 1);
          int h = cc >> 6,  d = cc & (DH - 1);
          if (MODE == 0)
            ((__bf16*)dst)[(((size_t)b * HH + h) * SS + s) * DH + d] = (__bf16)val;
          else
            ((__bf16*)dst)[(((size_t)b * HH + h) * DH + d) * SS + s] = (__bf16)val;
        }
      }
}

// ---------------------------------------------------------------------------
// Scores: attn[bh][q][k] = (Qh[bh] @ Kh[bh]^T) / 8, masked to -1e6 at k>=valid
// ---------------------------------------------------------------------------
__global__ __launch_bounds__(256) void scores_gemm(
    const __bf16* __restrict__ Qh, const __bf16* __restrict__ Kh,
    const int* __restrict__ valid, float* __restrict__ attn) {
  const int bh   = blockIdx.z;
  const int tid  = threadIdx.x;
  const int lane = tid & 31;
  const int wave = tid >> 5;
  const int wm   = wave >> 1;
  const int wn   = wave & 1;
  const int row0 = blockIdx.y * 128 + wm * 32;
  const int col0 = blockIdx.x * 128 + wn * 64;
  const int r    = lane & 15;
  const int hi   = lane >> 4;

  v8f acc[2][4];
#pragma unroll
  for (int i = 0; i < 2; ++i)
#pragma unroll
    for (int j = 0; j < 4; ++j) acc[i][j] = (v8f){0.f,0.f,0.f,0.f,0.f,0.f,0.f,0.f};

  const __bf16* ap0 = Qh + (size_t)bh * SS * DH + (size_t)(row0 + r) * DH + hi * 8;
  const __bf16* ap1 = ap0 + (size_t)16 * DH;
  const __bf16* bp0 = Kh + (size_t)bh * SS * DH + (size_t)(col0 + r) * DH + hi * 8;

#pragma unroll
  for (int k0 = 0; k0 < DH; k0 += 32) {
    v16bf a0 = frag_bf16(ap0 + k0);
    v16bf a1 = frag_bf16(ap1 + k0);
    v16bf b0 = frag_bf16(bp0 + k0);
    v16bf b1 = frag_bf16(bp0 + (size_t)16 * DH + k0);
    v16bf b2 = frag_bf16(bp0 + (size_t)32 * DH + k0);
    v16bf b3 = frag_bf16(bp0 + (size_t)48 * DH + k0);
    acc[0][0] = wmma_bf16(a0, b0, acc[0][0]);
    acc[0][1] = wmma_bf16(a0, b1, acc[0][1]);
    acc[0][2] = wmma_bf16(a0, b2, acc[0][2]);
    acc[0][3] = wmma_bf16(a0, b3, acc[0][3]);
    acc[1][0] = wmma_bf16(a1, b0, acc[1][0]);
    acc[1][1] = wmma_bf16(a1, b1, acc[1][1]);
    acc[1][2] = wmma_bf16(a1, b2, acc[1][2]);
    acc[1][3] = wmma_bf16(a1, b3, acc[1][3]);
  }

  const int vl = valid[bh >> 4];
  float* dst = attn + (size_t)bh * SS * SS;
#pragma unroll
  for (int i = 0; i < 2; ++i)
#pragma unroll
    for (int j = 0; j < 4; ++j)
#pragma unroll
      for (int e = 0; e < 8; ++e) {
        int q = row0 + i * 16 + hi * 8 + e;
        int k = col0 + j * 16 + r;
        float v = acc[i][j][e] * 0.125f;   // 1/sqrt(64)
        dst[(size_t)q * SS + k] = (k < vl) ? v : NEGI;
      }
}

// ---------------------------------------------------------------------------
// In-place row softmax over 2048-wide rows (65536 rows).
// ---------------------------------------------------------------------------
__global__ __launch_bounds__(256) void softmax_rows(float* __restrict__ attn) {
  __shared__ float red[256];
  float* p = attn + (size_t)blockIdx.x * SS;
  const int t = threadIdx.x;

  float vals[8];
  float m = -3.4e38f;
#pragma unroll
  for (int i = 0; i < 8; ++i) {
    vals[i] = p[t + i * 256];
    m = fmaxf(m, vals[i]);
  }
  red[t] = m;
  __syncthreads();
  for (int s = 128; s > 0; s >>= 1) {
    if (t < s) red[t] = fmaxf(red[t], red[t + s]);
    __syncthreads();
  }
  m = red[0];
  __syncthreads();

  float sum = 0.f;
#pragma unroll
  for (int i = 0; i < 8; ++i) {
    vals[i] = __expf(vals[i] - m);
    sum += vals[i];
  }
  red[t] = sum;
  __syncthreads();
  for (int s = 128; s > 0; s >>= 1) {
    if (t < s) red[t] += red[t + s];
    __syncthreads();
  }
  float inv = 1.0f / red[0];
#pragma unroll
  for (int i = 0; i < 8; ++i) p[t + i * 256] = vals[i] * inv;
}

// ---------------------------------------------------------------------------
// ctx = attn @ Vh, stored merged-head bf16: ctxm[(b*S+s)*D + h*64 + d]
// attn is f32 (converted to bf16 in the fragment load), VhT is per-head
// transposed so its rows are contiguous in K.
// Block: 128(M) x 64(N), wave tile 32x32.
// ---------------------------------------------------------------------------
__global__ __launch_bounds__(256) void attnv_gemm(
    const float* __restrict__ attn, const __bf16* __restrict__ VhT,
    __bf16* __restrict__ ctxm) {
  const int bh   = blockIdx.y;
  const int tid  = threadIdx.x;
  const int lane = tid & 31;
  const int wave = tid >> 5;
  const int wm   = wave >> 1;          // 0..3
  const int wn   = wave & 1;           // 0..1
  const int row0 = blockIdx.x * 128 + wm * 32;
  const int col0 = wn * 32;            // within head (0 or 32)
  const int r    = lane & 15;
  const int hi   = lane >> 4;

  v8f acc[2][2];
#pragma unroll
  for (int i = 0; i < 2; ++i)
#pragma unroll
    for (int j = 0; j < 2; ++j) acc[i][j] = (v8f){0.f,0.f,0.f,0.f,0.f,0.f,0.f,0.f};

  const float*  ap0 = attn + (size_t)bh * SS * SS + (size_t)(row0 + r) * SS + hi * 8;
  const float*  ap1 = ap0 + (size_t)16 * SS;
  const __bf16* bp0 = VhT + (size_t)bh * DH * SS + (size_t)(col0 + r) * SS + hi * 8;
  const __bf16* bp1 = bp0 + (size_t)16 * SS;

  for (int k0 = 0; k0 < SS; k0 += 32) {
    v16bf a0 = frag_f32(ap0 + k0);
    v16bf a1 = frag_f32(ap1 + k0);
    v16bf b0 = frag_bf16(bp0 + k0);
    v16bf b1 = frag_bf16(bp1 + k0);
    acc[0][0] = wmma_bf16(a0, b0, acc[0][0]);
    acc[0][1] = wmma_bf16(a0, b1, acc[0][1]);
    acc[1][0] = wmma_bf16(a1, b0, acc[1][0]);
    acc[1][1] = wmma_bf16(a1, b1, acc[1][1]);
  }

  const int b = bh >> 4, h = bh & (HH - 1);
#pragma unroll
  for (int i = 0; i < 2; ++i)
#pragma unroll
    for (int j = 0; j < 2; ++j)
#pragma unroll
      for (int e = 0; e < 8; ++e) {
        int s = row0 + i * 16 + hi * 8 + e;
        int d = col0 + j * 16 + r;                 // 0..63 within head
        ctxm[((size_t)b * SS + s) * DD + h * DH + d] = (__bf16)acc[i][j][e];
      }
}

// ---------------------------------------------------------------------------
// Host-side launch
// ---------------------------------------------------------------------------
extern "C" void kernel_launch(void* const* d_in, const int* in_sizes, int n_in,
                              void* d_out, int out_size, void* d_ws, size_t ws_size,
                              hipStream_t stream) {
  (void)in_sizes; (void)n_in; (void)out_size; (void)ws_size;

  const float* queries = (const float*)d_in[0];
  const float* keys    = (const float*)d_in[1];
  const float* values  = (const float*)d_in[2];
  const int*   valid   = (const int*)d_in[3];
  const float* W_q = (const float*)d_in[4];
  const float* b_q = (const float*)d_in[5];
  const float* W_k = (const float*)d_in[6];
  const float* b_k = (const float*)d_in[7];
  const float* W_v = (const float*)d_in[8];
  const float* b_v = (const float*)d_in[9];
  const float* W_o = (const float*)d_in[10];
  const float* b_o = (const float*)d_in[11];

  // Workspace layout (all bf16, 64 MB total)
  char* w = (char*)d_ws;
  const size_t XB = (size_t)MS * DD * 2;   // 8 MB per (B,S,D) bf16 tensor
  const size_t WB = (size_t)DD * DD * 2;   // 2 MB per weight
  __bf16* qb   = (__bf16*)w;  w += XB;
  __bf16* kb   = (__bf16*)w;  w += XB;
  __bf16* vb   = (__bf16*)w;  w += XB;
  __bf16* wqb  = (__bf16*)w;  w += WB;
  __bf16* wkb  = (__bf16*)w;  w += WB;
  __bf16* wvb  = (__bf16*)w;  w += WB;
  __bf16* wob  = (__bf16*)w;  w += WB;
  __bf16* Qh   = (__bf16*)w;  w += XB;   // (BH, S, 64)
  __bf16* Kh   = (__bf16*)w;  w += XB;   // (BH, S, 64)
  __bf16* VhT  = (__bf16*)w;  w += XB;   // (BH, 64, S)
  __bf16* ctxm = (__bf16*)w;  w += XB;   // (B, S, D) merged heads

  float* out  = (float*)d_out;
  float* attn = out + (size_t)MS * DD;   // (BH, S, S) f32, output #2

  const int NX = MS * DD;   // 4,194,304
  const int NW = DD * DD;   // 1,048,576

  // 1) f32 -> bf16
  mha_cvt_bf16<<<NX / 256, 256, 0, stream>>>(queries, qb, NX);
  mha_cvt_bf16<<<NX / 256, 256, 0, stream>>>(keys,    kb, NX);
  mha_cvt_bf16<<<NX / 256, 256, 0, stream>>>(values,  vb, NX);
  mha_cvt_bf16<<<NW / 256, 256, 0, stream>>>(W_q, wqb, NW);
  mha_cvt_bf16<<<NW / 256, 256, 0, stream>>>(W_k, wkb, NW);
  mha_cvt_bf16<<<NW / 256, 256, 0, stream>>>(W_v, wvb, NW);
  mha_cvt_bf16<<<NW / 256, 256, 0, stream>>>(W_o, wob, NW);

  // 2) projections (WMMA)
  dim3 gproj(DD / 128, MS / 128);   // (8, 32)
  proj_gemm<0><<<gproj, 256, 0, stream>>>(qb, wqb, b_q, Qh,  MS, DD, DD);
  proj_gemm<0><<<gproj, 256, 0, stream>>>(kb, wkb, b_k, Kh,  MS, DD, DD);
  proj_gemm<1><<<gproj, 256, 0, stream>>>(vb, wvb, b_v, VhT, MS, DD, DD);

  // 3) scaled+masked scores (WMMA) -> attn region of d_out
  scores_gemm<<<dim3(SS / 128, SS / 128, BB * HH), 256, 0, stream>>>(Qh, Kh, valid, attn);

  // 4) in-place row softmax
  softmax_rows<<<BB * HH * SS, 256, 0, stream>>>(attn);

  // 5) attn @ V (WMMA, f32->bf16 A fragments) -> merged-head context
  attnv_gemm<<<dim3(SS / 128, BB * HH), 256, 0, stream>>>(attn, VhT, ctxm);

  // 6) output projection (WMMA) -> out
  proj_gemm<2><<<gproj, 256, 0, stream>>>(ctxm, wob, b_o, out, MS, DD, DD);
}
